// CrossAttention_53798760350140
// MI455X (gfx1250) — compile-verified, matrix operand order checked by reading, not run on hardware
//
#include <hip/hip_runtime.h>
#include <hip/hip_bf16.h>
#include <math.h>

// ---------------------------------------------------------------------------
// Fused cross-attention + MLP + residual + LayerNorm for MI455X (gfx1250)
// bf16 WMMA (v_wmma_f32_16x16x32_bf16), f32 accumulate, TDM context staging.
// ---------------------------------------------------------------------------

typedef __attribute__((ext_vector_type(16))) __bf16 v16bf;
typedef __attribute__((ext_vector_type(8)))  float  v8f;

#define BB 8
#define DD 256
#define MM 1024
#define NN 4096            // H*W
#define MC 64              // context rows staged per chunk
#define NWAVES 4
#define TPB (NWAVES * 32)
#define STAGE_LD 264       // 256 + 8 pad (multiple of 8 for 16B-aligned frags)
#define CTXTP_LD 72        // 64 + 8 pad (multiple of 8)

// LDS layout (bytes) inside dynamic shared memory
#define LDS_CTX_RM 0
#define LDS_CTX_TP (LDS_CTX_RM + MC * DD * 2)                // +32768
#define LDS_P      (LDS_CTX_TP + DD * CTXTP_LD * 2)          // +36864
#define LDS_STA    (LDS_P + NWAVES * 16 * MC * 2)            // +8192
#define LDS_STB    (LDS_STA + NWAVES * 16 * STAGE_LD * 2)    // +33792
#define LDS_RED    (LDS_STB + NWAVES * 16 * STAGE_LD * 2)    // +33792
#define LDS_TOTAL  (LDS_RED + NWAVES * 2 * 4)                // = 145440 B

// Workspace layout (bytes)
#define XB_OFF  ((size_t)0)
#define CB_OFF  (XB_OFF + (size_t)BB * NN * DD * 2)          // 16 MB
#define W1B_OFF (CB_OFF + (size_t)BB * MM * DD * 2)          // +4 MB
#define W2B_OFF (W1B_OFF + (size_t)DD * DD * 2)
#define Y_OFF   (W2B_OFF + (size_t)DD * DD * 2)
#define ST_OFF  (Y_OFF + (size_t)BB * NN * DD * 4)           // ~54.8 MB total

// ---------------- TDM support (toolchain-portable) -------------------------
#if defined(__has_builtin)
#  if __has_builtin(__builtin_amdgcn_tensor_load_to_lds)
#    define HAVE_TDM 1
#  endif
#endif

#ifdef HAVE_TDM
typedef unsigned int u32x4 __attribute__((ext_vector_type(4)));
typedef int          i32x4 __attribute__((ext_vector_type(4)));
typedef int          i32x8 __attribute__((ext_vector_type(8)));

// 2D tile load: MC x DD bf16, tensor stride DD, into LDS at lds_off.
// D# fields per CDNA5 ISA 8.3/8.4 (count=1, type=2, data_size=2B).
__device__ __forceinline__ void tdm_load_ctx(unsigned lds_off, const void* gptr) {
  unsigned long long ga = (unsigned long long)(uintptr_t)gptr;
  u32x4 g0;
  g0.x = 1u;                                           // count=1, user mode
  g0.y = lds_off;                                      // lds_addr
  g0.z = (unsigned)ga;                                 // global_addr[31:0]
  g0.w = ((unsigned)(ga >> 32) & 0x01FFFFFFu) | (2u << 30);  // addr[56:32], type=2
  i32x8 g1;
  g1[0] = 1 << 16;                    // workgroup_mask=0, data_size=1 (2 bytes)
  g1[1] = (DD & 0xFFFF) << 16;        // tensor_dim0 lo16 -> bits[63:48]
  g1[2] = (MM & 0xFFFF) << 16;        // tensor_dim0 hi=0 | tensor_dim1 lo16
  g1[3] = (DD & 0xFFFF) << 16;        // tensor_dim1 hi=0 | tile_dim0 = DD
  g1[4] = MC & 0xFFFF;                // tile_dim1 = MC, tile_dim2 = 0
  g1[5] = DD;                         // tensor_dim0_stride lo32
  g1[6] = 0;                          // stride hi, dim1_stride lo (unused, 2D)
  g1[7] = 0;
  i32x4 z4 = {0, 0, 0, 0};
#if __has_include(<hip/amd_detail/amd_gfx1250_TDM.h>)
  i32x8 z8 = {0, 0, 0, 0, 0, 0, 0, 0};
  __builtin_amdgcn_tensor_load_to_lds(g0, g1, z4, z4, z8, 0);   // clang-23 form
#else
  __builtin_amdgcn_tensor_load_to_lds(g0, g1, z4, z4, 0);       // ROCm 7.2 form
#endif
}
#endif  // HAVE_TDM

__device__ __forceinline__ void wait_tensorcnt0() {
#if defined(__has_builtin)
#  if __has_builtin(__builtin_amdgcn_s_wait_tensorcnt)
  __builtin_amdgcn_s_wait_tensorcnt(0);
#  else
  asm volatile("s_wait_tensorcnt 0x0" ::: "memory");
#  endif
#else
  asm volatile("s_wait_tensorcnt 0x0" ::: "memory");
#endif
}

// ---------------------------------------------------------------------------
union FragU { uint4 u[2]; v16bf v; unsigned short s[16]; };

__device__ __forceinline__ unsigned short f2bf(float f) {
  unsigned int u = __float_as_uint(f);
  u += 0x7FFFu + ((u >> 16) & 1u);               // round-to-nearest-even
  return (unsigned short)(u >> 16);
}

// 16x32 bf16 A/B fragment: lane holds row/col (lane&15); halves are
// K = 8*hi + [0..7] and K = 16 + 8*hi + [0..7] (ISA 7.12.2 layout).
__device__ __forceinline__ v16bf ldfrag(const unsigned short* p) {
  FragU f;
  f.u[0] = *(const uint4*)(p);
  f.u[1] = *(const uint4*)(p + 16);
  return f.v;
}

__device__ __forceinline__ v8f vzero8() {
  v8f z = {0.f, 0.f, 0.f, 0.f, 0.f, 0.f, 0.f, 0.f};
  return z;
}

#define WMMA_BF16(A, B, C) \
  __builtin_amdgcn_wmma_f32_16x16x32_bf16(false, (A), false, (B), (short)0, (C), false, false)

// ---------------------------------------------------------------------------
// Kernel 0: f32 -> bf16 conversion + zero the LN stat accumulators
// ---------------------------------------------------------------------------
__global__ void convert_kernel(const float* __restrict__ x, const float* __restrict__ ctx,
                               const float* __restrict__ W1, const float* __restrict__ W2,
                               unsigned short* __restrict__ xb, unsigned short* __restrict__ cb,
                               unsigned short* __restrict__ w1b, unsigned short* __restrict__ w2b,
                               float* __restrict__ stats) {
  size_t i = (size_t)blockIdx.x * blockDim.x + threadIdx.x;
  if (i < (size_t)BB * NN * DD) xb[i] = f2bf(x[i]);
  if (i < (size_t)BB * MM * DD) cb[i] = f2bf(ctx[i]);
  if (i < (size_t)DD * DD) { w1b[i] = f2bf(W1[i]); w2b[i] = f2bf(W2[i]); }
  if (i < 32) stats[i] = 0.f;
}

// ---------------------------------------------------------------------------
// Kernel 1: fused flash-attention + 2-layer MLP + residual + LN-stat partials
// One wave owns a 16-row tile; 4 waves per block; 64 blocks per batch.
// ---------------------------------------------------------------------------
__global__ __launch_bounds__(TPB, 1)
void fused_attn_mlp_kernel(const float* __restrict__ x, const float* __restrict__ b1,
                           const float* __restrict__ b2,
                           const unsigned short* __restrict__ xb,
                           const unsigned short* __restrict__ cb,
                           const unsigned short* __restrict__ w1b,
                           const unsigned short* __restrict__ w2b,
                           float* __restrict__ y, float* __restrict__ stats) {
  extern __shared__ char lds[];
  unsigned short* ctx_rm = (unsigned short*)(lds + LDS_CTX_RM);
  unsigned short* ctx_tp = (unsigned short*)(lds + LDS_CTX_TP);

  const int tid  = threadIdx.x;
  const int wave = tid >> 5;
  const int lane = tid & 31;
  const int hi   = lane >> 4;     // half-wave select (K-chunk select in frags)
  const int ln   = lane & 15;     // row (A) / col (B,C,D) index

  unsigned short* Pb  = (unsigned short*)(lds + LDS_P)   + wave * 16 * MC;
  unsigned short* stA = (unsigned short*)(lds + LDS_STA) + wave * 16 * STAGE_LD;
  unsigned short* stB = (unsigned short*)(lds + LDS_STB) + wave * 16 * STAGE_LD;
  float* red = (float*)(lds + LDS_RED);

  const int batch = blockIdx.x >> 6;        // 64 blocks per batch
  const int btile = blockIdx.x & 63;
  const int row0  = btile * (NWAVES * 16) + wave * 16;

  // Stage this wave's 16x256 X tile (bf16) into its stage buffer: A-fragment
  // loads then come from short-latency LDS instead of hoarding 64 VGPRs of
  // global-loaded fragments (the round-1 register-pressure culprit).
  {
    const unsigned short* xsrc = xb + ((size_t)batch * NN + row0) * DD;
    for (int it = lane; it < 16 * (DD / 8); it += 32) {
      int r  = it >> 5;              // DD/8 == 32 uint4 per row
      int c8 = it & 31;
      *(uint4*)(stA + r * STAGE_LD + c8 * 8) = *(const uint4*)(xsrc + r * DD + c8 * 8);
    }
  }

  v8f acc[16];                               // O accumulator: 16 d-tiles x 16x16
  #pragma unroll
  for (int t = 0; t < 16; ++t) acc[t] = vzero8();
  float m_run[8], l_run[8];
  #pragma unroll
  for (int g = 0; g < 8; ++g) { m_run[g] = -1e30f; l_run[g] = 0.f; }

  // ---------------- flash attention over M in chunks of MC ----------------
  for (int c0 = 0; c0 < MM; c0 += MC) {
    __syncthreads();   // previous chunk fully consumed before overwrite
    const unsigned short* cbase = cb + ((size_t)batch * MM + c0) * DD;

#ifdef HAVE_TDM
    if (wave == 0) tdm_load_ctx((unsigned)LDS_CTX_RM, (const void*)cbase);  // row-major via TDM
#endif
    // Transposed copy (for PV B-fragments): 4-row groups -> packed b64 stores
    for (int it = tid; it < (MC / 4) * (DD / 8); it += TPB) {
      int r4 = it >> 5;              // 4-row group index
      int c8 = it & 31;
      const unsigned short* src = cbase + (size_t)(r4 * 4) * DD + c8 * 8;
      uint4 va = *(const uint4*)(src);
      uint4 vb = *(const uint4*)(src + DD);
      uint4 vc = *(const uint4*)(src + 2 * DD);
      uint4 vd = *(const uint4*)(src + 3 * DD);
#ifndef HAVE_TDM
      unsigned short* dst = ctx_rm + (r4 * 4) * DD + c8 * 8;
      *(uint4*)(dst)          = va;
      *(uint4*)(dst + DD)     = vb;
      *(uint4*)(dst + 2 * DD) = vc;
      *(uint4*)(dst + 3 * DD) = vd;
#endif
      const unsigned short* ea = (const unsigned short*)&va;
      const unsigned short* eb = (const unsigned short*)&vb;
      const unsigned short* ec = (const unsigned short*)&vc;
      const unsigned short* ed = (const unsigned short*)&vd;
      #pragma unroll
      for (int j = 0; j < 8; ++j) {
        unsigned long long pk = (unsigned long long)ea[j]
                              | ((unsigned long long)eb[j] << 16)
                              | ((unsigned long long)ec[j] << 32)
                              | ((unsigned long long)ed[j] << 48);
        *(unsigned long long*)(ctx_tp + (c8 * 8 + j) * CTXTP_LD + r4 * 4) = pk;
      }
    }
#ifdef HAVE_TDM
    if (wave == 0) wait_tensorcnt0();   // TDM DMA overlapped with transpose work
#endif
    if (c0 + MC < MM) __builtin_prefetch(cbase + (size_t)MC * DD, 0, 1);
    __syncthreads();

    // scores S[16 x MC] = X_tile[16 x 256] * Cchunk^T : 4 n-tiles, K=256
    v8f S[4];
    #pragma unroll
    for (int nt = 0; nt < 4; ++nt) S[nt] = vzero8();
    #pragma unroll
    for (int kb = 0; kb < 8; ++kb) {
      v16bf a = ldfrag(stA + ln * STAGE_LD + kb * 32 + 8 * hi);
      #pragma unroll
      for (int nt = 0; nt < 4; ++nt) {
        v16bf bm = ldfrag(ctx_rm + (nt * 16 + ln) * DD + kb * 32 + 8 * hi);
        S[nt] = WMMA_BF16(a, bm, S[nt]);
      }
    }

    // online softmax: row m = g + 8*hi lives across the 16 lanes of a half
    float scl[8];
    #pragma unroll
    for (int g = 0; g < 8; ++g) {
      float mx = S[0][g];
      #pragma unroll
      for (int nt = 1; nt < 4; ++nt) mx = fmaxf(mx, S[nt][g]);
      #pragma unroll
      for (int msk = 1; msk < 16; msk <<= 1) mx = fmaxf(mx, __shfl_xor(mx, msk));
      float mnew = fmaxf(m_run[g], mx);
      scl[g] = __expf(m_run[g] - mnew);
      float ssum = 0.f;
      #pragma unroll
      for (int nt = 0; nt < 4; ++nt) {
        float p = __expf(S[nt][g] - mnew);
        S[nt][g] = p;
        ssum += p;
      }
      #pragma unroll
      for (int msk = 1; msk < 16; msk <<= 1) ssum += __shfl_xor(ssum, msk);
      l_run[g] = l_run[g] * scl[g] + ssum;
      m_run[g] = mnew;
    }
    #pragma unroll
    for (int t = 0; t < 16; ++t) {
      #pragma unroll
      for (int g = 0; g < 8; ++g) acc[t][g] *= scl[g];
    }

    // stage P (D-layout -> A-layout needs cross-lane movement: use LDS)
    #pragma unroll
    for (int nt = 0; nt < 4; ++nt) {
      #pragma unroll
      for (int g = 0; g < 8; ++g)
        Pb[(g + 8 * hi) * MC + nt * 16 + ln] = f2bf(S[nt][g]);
    }
    // same-wave DS store->load stays in order (DScnt); no barrier needed.

    // O += P[16 x MC] * Cchunk[MC x 256] : 16 d-tiles, K = MC = 64
    #pragma unroll
    for (int kb = 0; kb < 2; ++kb) {
      v16bf a = ldfrag(Pb + ln * MC + kb * 32 + 8 * hi);
      #pragma unroll
      for (int t = 0; t < 16; ++t) {
        v16bf bm = ldfrag(ctx_tp + (t * 16 + ln) * CTXTP_LD + kb * 32 + 8 * hi);
        acc[t] = WMMA_BF16(a, bm, acc[t]);
      }
    }
  }

  // ---------------- epilogue: 1/l scale, MLP, residual, LN stats ----------
  float inv[8];
  #pragma unroll
  for (int g = 0; g < 8; ++g) inv[g] = 1.0f / l_run[g];
  #pragma unroll
  for (int t = 0; t < 16; ++t) {
    #pragma unroll
    for (int g = 0; g < 8; ++g)
      stA[(g + 8 * hi) * STAGE_LD + t * 16 + ln] = f2bf(acc[t][g] * inv[g]);
  }

  // layer 1: h = relu(O @ W1^T + b1); B-frag[k][n] = W1[n][k] (row-major read)
  #pragma unroll
  for (int nt = 0; nt < 16; ++nt) {
    v8f h = vzero8();
    #pragma unroll
    for (int kb = 0; kb < 8; ++kb) {
      v16bf a  = ldfrag(stA + ln * STAGE_LD + kb * 32 + 8 * hi);
      v16bf bm = ldfrag(w1b + (size_t)(nt * 16 + ln) * DD + kb * 32 + 8 * hi);
      h = WMMA_BF16(a, bm, h);
    }
    float bias = b1[nt * 16 + ln];
    #pragma unroll
    for (int g = 0; g < 8; ++g)
      stB[(g + 8 * hi) * STAGE_LD + nt * 16 + ln] = f2bf(fmaxf(h[g] + bias, 0.f));
  }

  // layer 2 + bias + residual + y store + per-batch sum/sumsq partials
  float psum = 0.f, psq = 0.f;
  #pragma unroll
  for (int nt = 0; nt < 16; ++nt) {
    v8f o = vzero8();
    #pragma unroll
    for (int kb = 0; kb < 8; ++kb) {
      v16bf a  = ldfrag(stB + ln * STAGE_LD + kb * 32 + 8 * hi);
      v16bf bm = ldfrag(w2b + (size_t)(nt * 16 + ln) * DD + kb * 32 + 8 * hi);
      o = WMMA_BF16(a, bm, o);
    }
    float bias = b2[nt * 16 + ln];
    #pragma unroll
    for (int g = 0; g < 8; ++g) {
      int row = row0 + g + 8 * hi;
      int col = nt * 16 + ln;
      size_t idx = ((size_t)batch * NN + row) * DD + col;
      float val = o[g] + bias + x[idx];
      y[idx] = val;
      psum += val;
      psq  += val * val;
    }
  }

  #pragma unroll
  for (int msk = 1; msk < 32; msk <<= 1) {
    psum += __shfl_xor(psum, msk);
    psq  += __shfl_xor(psq,  msk);
  }
  if (lane == 0) { red[wave * 2] = psum; red[wave * 2 + 1] = psq; }
  __syncthreads();
  if (tid == 0) {
    float s = 0.f, ss = 0.f;
    for (int w = 0; w < NWAVES; ++w) { s += red[w * 2]; ss += red[w * 2 + 1]; }
    atomicAdd(&stats[batch * 2], s);
    atomicAdd(&stats[batch * 2 + 1], ss);
  }
}

// ---------------------------------------------------------------------------
// Kernel 2: finalize per-batch mean / rstd
// ---------------------------------------------------------------------------
__global__ void finalize_stats_kernel(float* __restrict__ stats) {
  int b = threadIdx.x;
  if (b < BB) {
    const float n = (float)NN * (float)DD;
    float mean = stats[b * 2] / n;
    float var  = stats[b * 2 + 1] / n - mean * mean;
    stats[16 + b] = mean;
    stats[24 + b] = rsqrtf(var + 1e-5f);
  }
}

// ---------------------------------------------------------------------------
// Kernel 3: normalize y -> out (vectorized)
// ---------------------------------------------------------------------------
__global__ void normalize_kernel(const float4* __restrict__ y,
                                 const float* __restrict__ stats,
                                 float4* __restrict__ out) {
  size_t i = (size_t)blockIdx.x * blockDim.x + threadIdx.x;
  const size_t per_b = (size_t)NN * DD / 4;
  size_t b = i / per_b;
  float mean = stats[16 + b];
  float r    = stats[24 + b];
  float4 v = y[i];
  v.x = (v.x - mean) * r;
  v.y = (v.y - mean) * r;
  v.z = (v.z - mean) * r;
  v.w = (v.w - mean) * r;
  out[i] = v;
}

// ---------------------------------------------------------------------------
extern "C" void kernel_launch(void* const* d_in, const int* in_sizes, int n_in,
                              void* d_out, int out_size, void* d_ws, size_t ws_size,
                              hipStream_t stream) {
  (void)in_sizes; (void)n_in; (void)out_size; (void)ws_size;
  const float* x   = (const float*)d_in[0];
  const float* ctx = (const float*)d_in[1];
  const float* W1  = (const float*)d_in[2];
  const float* b1  = (const float*)d_in[3];
  const float* W2  = (const float*)d_in[4];
  const float* b2  = (const float*)d_in[5];
  float* out = (float*)d_out;

  char* ws = (char*)d_ws;   // requires ~55 MB workspace
  unsigned short* xb  = (unsigned short*)(ws + XB_OFF);
  unsigned short* cb  = (unsigned short*)(ws + CB_OFF);
  unsigned short* w1b = (unsigned short*)(ws + W1B_OFF);
  unsigned short* w2b = (unsigned short*)(ws + W2B_OFF);
  float* y     = (float*)(ws + Y_OFF);
  float* stats = (float*)(ws + ST_OFF);

  const int convN = BB * NN * DD;                       // dominant range
  convert_kernel<<<(convN + 255) / 256, 256, 0, stream>>>(
      x, ctx, W1, W2, xb, cb, w1b, w2b, stats);

  fused_attn_mlp_kernel<<<BB * 64, TPB, LDS_TOTAL, stream>>>(
      x, b1, b2, xb, cb, w1b, w2b, y, stats);

  finalize_stats_kernel<<<1, 32, 0, stream>>>(stats);

  const int vec4 = BB * NN * DD / 4;
  normalize_kernel<<<(vec4 + 255) / 256, 256, 0, stream>>>(
      (const float4*)y, stats, (float4*)out);
}
